// iWave1D_12884901888096
// MI455X (gfx1250) — matrix operand order repaired and checked by reading, not compiled, not run on hardware
//
#include <hip/hip_runtime.h>
#include <math.h>

// ---------------------------------------------------------------------------
// iWave 1D lifting (4 stages) fused for gfx1250 / MI455X.
// Each lift stage: sp = lift3tap(src); dst += sp + 25.6*( conv4( conv1..3 chain ) )
// Conv chain as implicit GEMM on v_wmma_f32_16x16x32_f16:
//   - weights pre-packed in LDS in WMMA A-fragment register layout
//   - conv1 input pre-packed as im2col patches (9 taps padded to 16 halves)
//   - residual t = c1 + conv3(t2) realized as an extra WMMA on the live patch
//   - biases pre-loaded into registers (no LDS access in masked epilogues)
// All hot-loop LDS traffic is 32B vector loads/16B vector stores.
// ---------------------------------------------------------------------------

typedef __attribute__((ext_vector_type(16))) _Float16 v16h;
typedef __attribute__((ext_vector_type(8)))  _Float16 v8h;
typedef __attribute__((ext_vector_type(8)))  float    v8f;

#define HH     1024   // half height (2048/2)
#define WTOT   1024
#define NBATCH 8

__device__ __forceinline__ v8f wmma16(v16h a, v16h b, v8f c) {
  // D = A(16x32) * B(32x16) + C, f32 accumulate
  return __builtin_amdgcn_wmma_f32_16x16x32_f16(false, a, false, b, (short)0, c,
                                                false, false);
}

// Branch-free tanh: tanh(x) = 1 - 2/(exp2(2*log2e*x)+1). v_exp_f32 + v_rcp_f32.
__device__ __forceinline__ float fast_tanh(float x) {
  float e = __builtin_amdgcn_exp2f(x * 2.8853900817779268f);
  return 1.0f - 2.0f * __builtin_amdgcn_rcpf(e + 1.0f);
}

// ---- Weight fragments prebuilt in LDS in a contiguous-K 16-bit A 16x32 layout:
// lane<16  : M=lane,    halves 0..15 -> K=0..15
// lane>=16 : M=lane-16, halves 0..15 -> K=16..31
// K-order for 16-ch convs: K = tap*16 + cin (taps 0..8; K>=144 exact zero).
// mode: 0 = 16->16 conv, 1 = conv1 (1 ch, K=tap), 2 = conv4 (only M=0 row)
__device__ __forceinline__ void build_frags(_Float16* __restrict__ dstf,
                                            const float* __restrict__ wg,
                                            int NS, int mode, int tid) {
  for (int idx = tid; idx < NS * 512; idx += 256) {
    int s    = idx >> 9;
    int r    = idx & 511;
    int lane = r >> 4;
    int i    = r & 15;
    int m    = lane & 15;
    int hiK  = (lane >> 4) * 16;
    int K    = 32 * s + i + hiK;
    float v = 0.0f;
    if (mode == 1) {
      if (K < 9) v = wg[m * 9 + K];
    } else if (mode == 0) {
      int tap = K >> 4, ci = K & 15;
      if (tap < 9) v = wg[(m * 16 + ci) * 9 + tap];
    } else {
      int tap = K >> 4, ci = K & 15;
      if (m == 0 && tap < 9) v = wg[ci * 9 + tap];
    }
    dstf[idx] = (_Float16)v;
  }
}

__device__ __forceinline__ v16h load_frag(const _Float16* __restrict__ f, int s, int lane) {
  return *(const v16h*)(f + (s * 32 + lane) * 16);   // 32B aligned vector load
}

// ---- B fragment gather (contiguous-K B 32x16 layout, matching the A fragments):
// lane<16 : column N=lane,    halves -> K=0..15  (tap 2s,   cin 0..15)
// lane>=16: column N=lane-16, halves -> K=16..31 (tap 2s+1, cin 0..15)
// tap clamped: garbage columns hit exact-zero A rows (finite * 0 = 0).
__device__ __forceinline__ v16h gather_B16(const _Float16* __restrict__ in, int IW,
                                           int s, int lane, int orow, int ocol) {
  int tap = 2 * s + (lane >> 4);
  tap = (tap < 9) ? tap : 8;
  int ty = tap / 3, tx = tap - ty * 3;
  return *(const v16h*)(in + (((orow + ty) * IW) + (ocol + tx)) * 16);
}

__global__ void __launch_bounds__(256)
lift_step_kernel(const float* __restrict__ src, float* __restrict__ dst,
                 const float* __restrict__ lw,
                 const float* __restrict__ w1, const float* __restrict__ b1,
                 const float* __restrict__ w2, const float* __restrict__ b2,
                 const float* __restrict__ w3, const float* __restrict__ b3,
                 const float* __restrict__ w4, const float* __restrict__ b4) {
  // LDS staging (~62.9 KB)
  __shared__ __align__(16) float    s_sp   [24 * 24];      // sp f32 (residual+patch src)
  __shared__ __align__(32) _Float16 s_patch[484 * 16];     // im2col of sp/256, 22x22 c1 region
  __shared__ __align__(32) _Float16 s_t1   [484 * 16];     // tanh(c1) 22-wide; reused for t 18-wide
  __shared__ __align__(32) _Float16 s_t2   [400 * 16];     // tanh(conv2), 20-wide
  __shared__ __align__(32) _Float16 s_w1f  [1 * 512];      // A-fragments
  __shared__ __align__(32) _Float16 s_w2f  [5 * 512];
  __shared__ __align__(32) _Float16 s_w3f  [5 * 512];
  __shared__ __align__(32) _Float16 s_w4f  [5 * 512];
  __shared__ __align__(32) float s_b1[16], s_b2[16], s_b3[16];
  __shared__ float s_b4[1], s_lift[3];

  const int tid  = threadIdx.x;
  const int lane = tid & 31;
  const int wid  = tid >> 5;
  const int mb   = (lane >> 4) * 8;       // M-row base held by this lane
  const int bh0  = blockIdx.y * 16;
  const int bw0  = blockIdx.x * 16;
  const int sh0  = bh0 - 4;
  const int sw0  = bw0 - 4;

  float* dst_b = dst + (size_t)blockIdx.z * HH * WTOT;
  // Warm L2 for the += read-modify-write in conv4 (global_prefetch_b8).
  if (tid < 16)
    __builtin_prefetch((const void*)(dst_b + (size_t)(bh0 + tid) * WTOT + bw0), 0, 0);

  // ---- weight fragments + biases -> LDS ----
  build_frags(s_w1f, w1, 1, 1, tid);
  build_frags(s_w2f, w2, 5, 0, tid);
  build_frags(s_w3f, w3, 5, 0, tid);
  build_frags(s_w4f, w4, 5, 2, tid);
  if (tid < 16) { s_b1[tid] = b1[tid]; s_b2[tid] = b2[tid]; s_b3[tid] = b3[tid]; }
  if (tid == 0) { s_b4[0] = b4[0]; s_lift[0] = lw[0]; s_lift[1] = lw[1]; s_lift[2] = lw[2]; }
  __syncthreads();

  // Per-lane bias vectors in registers: nothing in the masked epilogues touches LDS.
  const v8f bv1  = *(const v8f*)(s_b1 + mb);
  const v8f bv2  = *(const v8f*)(s_b2 + mb);
  const v8f bv13 = bv1 + *(const v8f*)(s_b3 + mb);
  const float b4r = s_b4[0];

  // ---- sp tile 24x24 (lift 3-tap along H, reflect at image edge, 0 outside) ----
  const float* src_b = src + (size_t)blockIdx.z * HH * WTOT;
  for (int i = tid; i < 576; i += 256) {
    int r = i / 24, c = i % 24;
    int h = sh0 + r, w = sw0 + c;
    float v = 0.0f;
    if (h >= 0 && h < HH && w >= 0 && w < WTOT) {
      int hm = h - 1; if (hm < 0) hm = -hm;
      int hp = h + 1; if (hp >= HH) hp = 2 * HH - 2 - hp;
      v = s_lift[0] * src_b[(size_t)hm * WTOT + w]
        + s_lift[1] * src_b[(size_t)h  * WTOT + w]
        + s_lift[2] * src_b[(size_t)hp * WTOT + w];
    }
    s_sp[i] = v;
  }
  __syncthreads();

  // ---- im2col patches for conv1: 22x22 pixels x 16 halves (taps 0..8 + pad) ----
  for (int p = tid; p < 484; p += 256) {
    int orow = p / 22, ocol = p % 22;
    union { v16h v; _Float16 h[16]; } u;
#pragma unroll
    for (int i = 0; i < 16; ++i) {
      int tap = (i < 9) ? i : 8;                    // pad halves: finite, hits zero A
      int ty = tap / 3, tx = tap - ty * 3;
      u.h[i] = (_Float16)(s_sp[(orow + ty) * 24 + (ocol + tx)] * (1.0f / 256.0f));
    }
    *(v16h*)(s_patch + p * 16) = u.v;
  }
  __syncthreads();

  // conv1 A-fragment stays live through conv3 (residual re-GEMM).
  const v16h a1 = load_frag(s_w1f, 0, lane);

  // ---- conv1 (1->16) on 22x22: t1 = tanh(conv1+b1), masked outside image ----
  {
    for (int g = wid; g < 31; g += 8) {
      const int pt = g * 16 + (lane & 15);
      const int p  = (pt < 484) ? pt : 483;
      const int orow = p / 22, ocol = p % 22;
      v8f c = {};
      c = wmma16(a1, *(const v16h*)(s_patch + p * 16), c);
      const int habs = sh0 + 1 + orow, wabs = sw0 + 1 + ocol;
      const bool inimg = (habs >= 0) && (habs < HH) && (wabs >= 0) && (wabs < WTOT);
      if (pt < 484) {
        union { v8h v; _Float16 h[8]; } ut;
#pragma unroll
        for (int i = 0; i < 8; ++i) {
          float v = fast_tanh(c[i] + bv1[i]);
          ut.h[i] = inimg ? (_Float16)v : (_Float16)0.0f;
        }
        *(v8h*)(s_t1 + p * 16 + mb) = ut.v;
      }
    }
  }
  __syncthreads();

  // ---- conv2 (16->16) on 20x20: t2 = tanh(conv2(t1)+b2), masked ----
  {
    v16h a[5];
#pragma unroll
    for (int s = 0; s < 5; ++s) a[s] = load_frag(s_w2f, s, lane);
    for (int g = wid; g < 25; g += 8) {   // 400/16 exact
      const int p = g * 16 + (lane & 15);
      const int orow = p / 20, ocol = p % 20;
      v8f c = {};
#pragma unroll
      for (int s = 0; s < 5; ++s)
        c = wmma16(a[s], gather_B16(s_t1, 22, s, lane, orow, ocol), c);
      const int habs = sh0 + 2 + orow, wabs = sw0 + 2 + ocol;
      const bool inimg = (habs >= 0) && (habs < HH) && (wabs >= 0) && (wabs < WTOT);
      union { v8h v; _Float16 h[8]; } u;
#pragma unroll
      for (int i = 0; i < 8; ++i) {
        float v = fast_tanh(c[i] + bv2[i]);
        u.h[i] = inimg ? (_Float16)v : (_Float16)0.0f;
      }
      *(v8h*)(s_t2 + p * 16 + mb) = u.v;
    }
  }
  __syncthreads();

  // ---- conv3 (16->16) on 18x18: t = conv1(patch) + conv3(t2) + b1 + b3 ----
  // residual c1 recomputed in the f32 accumulator via one extra WMMA on the patch
  {
    v16h a[5];
#pragma unroll
    for (int s = 0; s < 5; ++s) a[s] = load_frag(s_w3f, s, lane);
    for (int g = wid; g < 21; g += 8) {
      const int pt = g * 16 + (lane & 15);
      const int p  = (pt < 324) ? pt : 323;
      const int orow = p / 18, ocol = p % 18;
      v8f c = {};
#pragma unroll
      for (int s = 0; s < 5; ++s)
        c = wmma16(a[s], gather_B16(s_t2, 20, s, lane, orow, ocol), c);
      const int q = (orow + 2) * 22 + (ocol + 2);           // c1-region pixel
      c = wmma16(a1, *(const v16h*)(s_patch + q * 16), c);  // + c1 residual
      const int habs = sh0 + 3 + orow, wabs = sw0 + 3 + ocol;
      const bool inimg = (habs >= 0) && (habs < HH) && (wabs >= 0) && (wabs < WTOT);
      if (pt < 324) {
        union { v8h v; _Float16 h[8]; } u;
#pragma unroll
        for (int i = 0; i < 8; ++i) {
          float v = c[i] + bv13[i];
          u.h[i] = inimg ? (_Float16)v : (_Float16)0.0f;
        }
        *(v8h*)(s_t1 + p * 16 + mb) = u.v;   // t, compact 18-wide layout
      }
    }
  }
  __syncthreads();

  // ---- conv4 (16->1) on 16x16: dst += sp + 25.6*(conv4(t)+b4) ----
  {
    v16h a[5];
#pragma unroll
    for (int s = 0; s < 5; ++s) a[s] = load_frag(s_w4f, s, lane);
    for (int g = wid; g < 16; g += 8) {
      const int p = g * 16 + (lane & 15);
      const int orow = p >> 4, ocol = p & 15;
      v8f c = {};
#pragma unroll
      for (int s = 0; s < 5; ++s)
        c = wmma16(a[s], gather_B16(s_t1, 18, s, lane, orow, ocol), c);
      if (lane < 16) {  // row M=0 lives in c[0] of lanes 0..15
        float nnraw = c[0] + b4r;
        float res   = s_sp[(orow + 4) * 24 + (ocol + 4)] + 25.6f * nnraw;
        size_t idx = (size_t)(bh0 + orow) * WTOT + (size_t)(bw0 + ocol);
        dst_b[idx] += res;
      }
    }
  }
}

__global__ void deinterleave_kernel(const float* __restrict__ x,
                                    float* __restrict__ xe, float* __restrict__ xo) {
  size_t i = (size_t)blockIdx.x * blockDim.x + threadIdx.x;
  const size_t n = (size_t)NBATCH * HH * WTOT;
  if (i >= n) return;
  size_t w = i % WTOT;
  size_t t = i / WTOT;
  size_t h = t % HH;
  size_t b = t / HH;
  const float* xb = x + (b * (size_t)(2 * HH) + 2 * h) * WTOT;
  xe[i] = xb[w];
  xo[i] = xb[WTOT + w];
}

__global__ void finalize_kernel(const float* __restrict__ xe, const float* __restrict__ xo,
                                const float* __restrict__ sl, const float* __restrict__ sh,
                                float* __restrict__ out) {
  size_t i = (size_t)blockIdx.x * blockDim.x + threadIdx.x;
  const size_t n = (size_t)NBATCH * HH * WTOT;
  if (i >= n) return;
  out[i]     = xe[i] * sl[0];
  out[n + i] = xo[i] * sh[0];
}

extern "C" void kernel_launch(void* const* d_in, const int* in_sizes, int n_in,
                              void* d_out, int out_size, void* d_ws, size_t ws_size,
                              hipStream_t stream) {
  (void)in_sizes; (void)n_in; (void)out_size; (void)ws_size;
  const float* x  = (const float*)d_in[0];
  const float* lw = (const float*)d_in[1];
  const float* w1 = (const float*)d_in[2];
  const float* b1 = (const float*)d_in[3];
  const float* w2 = (const float*)d_in[4];
  const float* b2 = (const float*)d_in[5];
  const float* w3 = (const float*)d_in[6];
  const float* b3 = (const float*)d_in[7];
  const float* w4 = (const float*)d_in[8];
  const float* b4 = (const float*)d_in[9];
  const float* sl = (const float*)d_in[10];
  const float* sh = (const float*)d_in[11];
  float* out = (float*)d_out;

  const size_t half = (size_t)NBATCH * HH * WTOT;  // 8 Mi elements
  float* xe = (float*)d_ws;
  float* xo = xe + half;

  {
    int thr = 256;
    deinterleave_kernel<<<dim3((unsigned)((half + thr - 1) / thr)), thr, 0, stream>>>(x, xe, xo);
  }

  dim3 grid(WTOT / 16, HH / 16, NBATCH);
  // step 0: x_o += lift(x_e)
  lift_step_kernel<<<grid, 256, 0, stream>>>(xe, xo, lw + 0,
      w1 + 0,   b1 + 0,  w2 + 0,    b2 + 0,  w3 + 0,    b3 + 0,  w4 + 0,   b4 + 0);
  // step 1: x_e += lift(x_o)
  lift_step_kernel<<<grid, 256, 0, stream>>>(xo, xe, lw + 3,
      w1 + 144, b1 + 16, w2 + 2304, b2 + 16, w3 + 2304, b3 + 16, w4 + 144, b4 + 1);
  // step 2: x_o += lift(x_e)
  lift_step_kernel<<<grid, 256, 0, stream>>>(xe, xo, lw + 6,
      w1 + 288, b1 + 32, w2 + 4608, b2 + 32, w3 + 4608, b3 + 32, w4 + 288, b4 + 2);
  // step 3: x_e += lift(x_o)
  lift_step_kernel<<<grid, 256, 0, stream>>>(xo, xe, lw + 9,
      w1 + 432, b1 + 48, w2 + 6912, b2 + 48, w3 + 6912, b3 + 48, w4 + 432, b4 + 3);

  {
    int thr = 256;
    finalize_kernel<<<dim3((unsigned)((half + thr - 1) / thr)), thr, 0, stream>>>(xe, xo, sl, sh, out);
  }
}
